// SNN_53618371723788
// MI455X (gfx1250) — matrix operand order repaired and checked by reading, not compiled, not run on hardware
//
#include <hip/hip_runtime.h>

typedef __attribute__((ext_vector_type(16))) _Float16 v16h;
typedef __attribute__((ext_vector_type(8)))  _Float16 v8h;
typedef __attribute__((ext_vector_type(8)))  float    v8f;

#define TT 24
#define MT 16          // batch rows per workgroup
#define NTHREADS 256   // 8 waves (wave32)

__device__ __forceinline__ float sigf(float v) {
    return 1.0f / (1.0f + __expf(-v));
}
__device__ __forceinline__ float tanh_fast(float v) {
    float e = __expf(2.0f * v);
    return (e - 1.0f) / (e + 1.0f);
}

// A-fragment loader: 16x32 f16 A tile, contiguous k order (same as B fragments).
// lane<16: elems 0..15 = K(ks*32+0..15)
// lane>=16: elems 0..15 = K(ks*32+16..31)
__device__ __forceinline__ v16h load_afrag(const _Float16* rowbase, int ks, int lhi) {
    const _Float16* p0 = rowbase + ks * 32 + lhi * 16;
    v8h lo = *(const v8h*)p0;
    v8h hi = *(const v8h*)(p0 + 8);
    return __builtin_shufflevector(lo, hi, 0,1,2,3,4,5,6,7,8,9,10,11,12,13,14,15);
}

__global__ __launch_bounds__(NTHREADS) void snn_slstm_fused(
    const float* __restrict__ x,
    const float* __restrict__ Wih1, const float* __restrict__ Whh1,
    const float* __restrict__ bih1, const float* __restrict__ bhh1,
    const float* __restrict__ thr1p,
    const float* __restrict__ Wih2, const float* __restrict__ Whh2,
    const float* __restrict__ bih2, const float* __restrict__ bhh2,
    const float* __restrict__ thr2p,
    const float* __restrict__ Wout, const float* __restrict__ bout,
    float* __restrict__ out)
{
    __shared__ _Float16 sW1[256 * 64];    // Whh1[n][k]            (32 KB)
    __shared__ _Float16 sW2[256 * 128];   // [Wih2 | Whh2][n][k]   (64 KB)
    __shared__ float    sB1[256], sB2[256], sWi1[256];
    __shared__ float    sX[MT * TT];
    __shared__ float    sG[MT * 256];     // gate scratch          (16 KB)
    __shared__ _Float16 sA1[MT * 64];     // mem1 f16 A-tile
    __shared__ _Float16 sA2[MT * 128];    // [spk1 | mem2] f16 A-tile
    __shared__ float    sRed[MT * 2];

    const int tid    = threadIdx.x;
    const int wgbase = blockIdx.x * MT;

    // ---- stage weights (f32 -> f16), biases, inputs into LDS ----
    for (int i = tid; i < 256 * 64; i += NTHREADS) sW1[i] = (_Float16)Whh1[i];
    for (int i = tid; i < 256 * 128; i += NTHREADS) {
        int n = i >> 7, k = i & 127;
        float v = (k < 64) ? Wih2[n * 64 + k] : Whh2[n * 64 + (k - 64)];
        sW2[i] = (_Float16)v;
    }
    if (tid < 256) {
        sB1[tid]  = bih1[tid] + bhh1[tid];
        sB2[tid]  = bih2[tid] + bhh2[tid];
        sWi1[tid] = Wih1[tid];
    }
    for (int i = tid; i < MT * TT; i += NTHREADS)
        sX[i] = x[(size_t)(wgbase + i / TT) * TT + (i % TT)];
    for (int i = tid; i < MT * 64; i += NTHREADS) {
        sA1[i] = (_Float16)0.f;                                // mem1(-1) = 0
        sA2[(i >> 6) * 128 + 64 + (i & 63)] = (_Float16)0.f;   // mem2(-1) = 0
    }
    if (tid < MT * 2) sRed[tid] = 0.f;
    __syncthreads();

    // ---- wave/lane indexing for WMMA ----
    const int wv  = tid >> 5;
    const int l   = tid & 31;
    const int lhi = l >> 4, llo = l & 15;

    // ---- preload ALL B fragments into registers (weights are loop-invariant) ----
    // B 32x16 f16 layout: lane = N column (llo); 16 contiguous K halves per lane,
    // lanes 16..31 take K+16. Row-major W[n][k] gives contiguity for free.
    v16h b1f[2][2];
    v16h b2f[2][4];
    #pragma unroll
    for (int p = 0; p < 2; ++p) {
        int n = wv * 32 + p * 16 + llo;
        #pragma unroll
        for (int ks = 0; ks < 2; ++ks) {
            const _Float16* bp = sW1 + n * 64 + ks * 32 + lhi * 16;
            v8h lo = *(const v8h*)bp, hi = *(const v8h*)(bp + 8);
            b1f[p][ks] = __builtin_shufflevector(lo, hi, 0,1,2,3,4,5,6,7,8,9,10,11,12,13,14,15);
        }
        #pragma unroll
        for (int ks = 0; ks < 4; ++ks) {
            const _Float16* bp = sW2 + n * 128 + ks * 32 + lhi * 16;
            v8h lo = *(const v8h*)bp, hi = *(const v8h*)(bp + 8);
            b2f[p][ks] = __builtin_shufflevector(lo, hi, 0,1,2,3,4,5,6,7,8,9,10,11,12,13,14,15);
        }
    }

    // ---- per-thread elementwise ownership: 4 (m,h) pairs ----
    const int eh  = tid & 63;   // neuron h
    const int emb = tid >> 6;   // m = emb + 4*j
    float s1[4] = {0,0,0,0}, m1[4] = {0,0,0,0};
    float s2[4] = {0,0,0,0}, m2[4] = {0,0,0,0};
    float o0[4] = {0,0,0,0}, o1a[4] = {0,0,0,0};
    const float thr1 = thr1p[0], thr2 = thr2p[0];

    // layer-2 GEMM: gates2 = [spk1|mem2](16x128) @ [Wih2;Whh2](128x256) + b2
    auto gemm2 = [&]() {
        const _Float16* arow = sA2 + llo * 128;
        v16h af[4];
        #pragma unroll
        for (int ks = 0; ks < 4; ++ks) af[ks] = load_afrag(arow, ks, lhi);
        #pragma unroll
        for (int p = 0; p < 2; ++p) {
            int n = wv * 32 + p * 16 + llo;
            float bb = sB2[n];
            v8f c;
            #pragma unroll
            for (int r = 0; r < 8; ++r) c[r] = bb;
            #pragma unroll
            for (int ks = 0; ks < 4; ++ks)
                c = __builtin_amdgcn_wmma_f32_16x16x32_f16(false, af[ks], false, b2f[p][ks],
                                                           (short)0, c, false, false);
            #pragma unroll
            for (int r = 0; r < 8; ++r)
                sG[(r + 8 * lhi) * 256 + n] = c[r];
        }
    };

    for (int t = 0; t < TT; ++t) {
        // ---- layer-1 GEMM: gates1 = mem1(16x64)@Whh1^T + x*Wih1 + b1 ----
        {
            const _Float16* arow = sA1 + llo * 64;
            v16h a0  = load_afrag(arow, 0, lhi);
            v16h a1v = load_afrag(arow, 1, lhi);
            #pragma unroll
            for (int p = 0; p < 2; ++p) {
                int n = wv * 32 + p * 16 + llo;
                float bb = sB1[n], wi = sWi1[n];
                v8f c;
                #pragma unroll
                for (int r = 0; r < 8; ++r)           // fold x*Wih1 rank-1 term into acc init
                    c[r] = fmaf(sX[(lhi * 8 + r) * TT + t], wi, bb);
                c = __builtin_amdgcn_wmma_f32_16x16x32_f16(false, a0,  false, b1f[p][0], (short)0, c, false, false);
                c = __builtin_amdgcn_wmma_f32_16x16x32_f16(false, a1v, false, b1f[p][1], (short)0, c, false, false);
                #pragma unroll
                for (int r = 0; r < 8; ++r)
                    sG[(r + 8 * lhi) * 256 + n] = c[r];
            }
        }
        __syncthreads();

        // ---- layer-1 elementwise ----
        #pragma unroll
        for (int j = 0; j < 4; ++j) {
            int m = emb + 4 * j;
            const float* g = sG + m * 256 + eh;
            float gi = g[0], gf = g[64], gg = g[128], go = g[192];
            float rst = (m1[j] - thr1 > 0.f) ? thr1 : 0.f;   // reset uses pre-update mem
            s1[j] = sigf(gf) * s1[j] + sigf(gi) * tanh_fast(gg);
            float mn = sigf(go) * tanh_fast(s1[j]) - rst;
            m1[j] = mn;
            float spk = (mn - thr1 > 0.f) ? 1.f : 0.f;
            sA1[m * 64 + eh]  = (_Float16)mn;   // mem1 -> next step's A1
            sA2[m * 128 + eh] = (_Float16)spk;  // spk1 -> layer-2 A (K 0..63)
        }
        __syncthreads();

        // ---- layer-2 GEMM ----
        gemm2();
        __syncthreads();

        // ---- layer-2 elementwise + fused Wout projection ----
        #pragma unroll
        for (int j = 0; j < 4; ++j) {
            int m = emb + 4 * j;
            const float* g = sG + m * 256 + eh;
            float gi = g[0], gf = g[64], gg = g[128], go = g[192];
            float rst = (m2[j] - thr2 > 0.f) ? thr2 : 0.f;
            s2[j] = sigf(gf) * s2[j] + sigf(gi) * tanh_fast(gg);
            float mn = sigf(go) * tanh_fast(s2[j]) - rst;
            m2[j] = mn;
            float spk = (mn - thr2 > 0.f) ? 1.f : 0.f;
            o0[j]  = fmaf(spk, Wout[t * 64 + eh],        o0[j]);
            o1a[j] = fmaf(spk, Wout[1664 + t * 64 + eh], o1a[j]);
            sA2[m * 128 + 64 + eh] = (_Float16)mn;       // mem2 -> layer-2 A (K 64..127)
        }
        __syncthreads();
    }

    // ---- extra step: slstm2(mem1_final) ----
    #pragma unroll
    for (int j = 0; j < 4; ++j) {
        int m = emb + 4 * j;
        sA2[m * 128 + eh] = (_Float16)m1[j];
    }
    __syncthreads();
    gemm2();
    __syncthreads();
    #pragma unroll
    for (int j = 0; j < 4; ++j) {
        int m = emb + 4 * j;
        const float* g = sG + m * 256 + eh;
        float gi = g[0], gf = g[64], gg = g[128], go = g[192];
        float rst = (m2[j] - thr2 > 0.f) ? thr2 : 0.f;
        s2[j] = sigf(gf) * s2[j] + sigf(gi) * tanh_fast(gg);
        float mn = sigf(go) * tanh_fast(s2[j]) - rst;
        float spk = (mn - thr2 > 0.f) ? 1.f : 0.f;
        // spk2_last -> cols 1536+h ; mem2_final -> cols 1600+h
        o0[j]  += spk * Wout[1536 + eh]        + mn * Wout[1600 + eh];
        o1a[j] += spk * Wout[1664 + 1536 + eh] + mn * Wout[1664 + 1600 + eh];
    }

    // ---- per-row reduction of fused output ----
    #pragma unroll
    for (int j = 0; j < 4; ++j) {
        int m = emb + 4 * j;
        atomicAdd(&sRed[m * 2 + 0], o0[j]);
        atomicAdd(&sRed[m * 2 + 1], o1a[j]);
    }
    __syncthreads();
    if (tid < MT * 2) {
        int m = tid >> 1, c = tid & 1;
        out[(size_t)(wgbase + m) * 2 + c] = sRed[tid] + bout[c];
    }
}

extern "C" void kernel_launch(void* const* d_in, const int* in_sizes, int n_in,
                              void* d_out, int out_size, void* d_ws, size_t ws_size,
                              hipStream_t stream) {
    const float* x    = (const float*)d_in[0];
    const float* Wih1 = (const float*)d_in[1];
    const float* Whh1 = (const float*)d_in[2];
    const float* bih1 = (const float*)d_in[3];
    const float* bhh1 = (const float*)d_in[4];
    const float* thr1 = (const float*)d_in[5];
    const float* Wih2 = (const float*)d_in[6];
    const float* Whh2 = (const float*)d_in[7];
    const float* bih2 = (const float*)d_in[8];
    const float* bhh2 = (const float*)d_in[9];
    const float* thr2 = (const float*)d_in[10];
    const float* Wout = (const float*)d_in[11];
    const float* bout = (const float*)d_in[12];
    float* out = (float*)d_out;

    int B = in_sizes[0] / TT;       // 65536
    int nblocks = B / MT;           // 4096 workgroups, 16 batch rows each
    hipLaunchKernelGGL(snn_slstm_fused, dim3(nblocks), dim3(NTHREADS), 0, stream,
                       x, Wih1, Whh1, bih1, bhh1, thr1,
                       Wih2, Whh2, bih2, bhh2, thr2, Wout, bout, out);
}
